// SAModule_23278722744324
// MI455X (gfx1250) — matrix operand order repaired and checked by reading, not compiled, not run on hardware
//
#include <hip/hip_runtime.h>

typedef __attribute__((ext_vector_type(16))) _Float16 v16h;
typedef __attribute__((ext_vector_type(8)))  float    v8f;

#define N_PTS   8192
#define PER_B   2048
#define C_IN    16
#define H1_DIM  64
#define H2_DIM  128
#define KNBR    64
#define R2_VAL  0.49f
#define CAP     1024

static __device__ __forceinline__ unsigned pk(float a, float b) {
  union { _Float16 h[2]; unsigned u; } z;
  z.h[0] = (_Float16)a; z.h[1] = (_Float16)b;
  return z.u;
}

// ---------------------------------------------------------------------------
// Phase 0: convert weights to f16 *pre-swizzled into WMMA B-fragment layout*.
// Fragment f stored as [lane 0..31][dword 0..7]; lane (hi,l16), dword v holds
// the K pair (kk, kk+1) with kk = hi*8 + 2(v&3) + 16(v>>2) for column l16 of the tile.
//   W1x: 4 N-tiles of W1 [K=19 padded to 32][64]   -> 4*256 dwords (4 KB)
//   W2x: 8 N-tiles x 2 K-steps of W2 [64][128]     -> 16*256 dwords (16 KB)
// ---------------------------------------------------------------------------
__global__ void prep_weights(const float* __restrict__ W1,
                             const float* __restrict__ W2,
                             unsigned* __restrict__ W1x,
                             unsigned* __restrict__ W2x) {
  int t = blockIdx.x * blockDim.x + threadIdx.x;
  if (t < 4 * 256) {
    int f = t >> 8, r = t & 255;
    int lane = r >> 3, v = r & 7;
    int hi = lane >> 4, l16 = lane & 15;
    int kk  = hi * 8 + 2 * (v & 3) + (v >> 2) * 16;
    int col = f * 16 + l16;
    float a = (kk     < C_IN + 3) ? W1[kk * H1_DIM + col]       : 0.f;
    float b = (kk + 1 < C_IN + 3) ? W1[(kk + 1) * H1_DIM + col] : 0.f;
    W1x[t] = pk(a, b);
  } else {
    int u = t - 1024;
    if (u < 16 * 256) {
      int f = u >> 8, r = u & 255;
      int lane = r >> 3, v = r & 7;
      int hi = lane >> 4, l16 = lane & 15;
      int nt = f >> 1, ks = f & 1;
      int kk  = ks * 32 + hi * 8 + 2 * (v & 3) + (v >> 2) * 16;
      int col = nt * 16 + l16;
      W2x[u] = pk(W2[kk * H2_DIM + col], W2[(kk + 1) * H2_DIM + col]);
    }
  }
}

// ---------------------------------------------------------------------------
// Phase 1: radius ball query + top-K (one workgroup per point).
// key = (f32bits(d2) << 32) | j -> unsigned sort == (d2, idx) ascending,
// matching lax.top_k's stable tie-break. Max-aggregation downstream is
// order-invariant, so atomic-append order does not affect d_out.
// ---------------------------------------------------------------------------
__global__ __launch_bounds__(256) void radius_topk(
    const float* __restrict__ pos,
    int* __restrict__ nbr, int* __restrict__ cnt) {
  __shared__ unsigned long long keys[CAP];
  __shared__ int scount;
  int i = blockIdx.x;
  int t = threadIdx.x;
  if (t == 0) scount = 0;
  __syncthreads();

  int base = (i >> 11) << 11;  // batch = repeat(arange(4), 2048), sorted
  float px = pos[3 * i], py = pos[3 * i + 1], pz = pos[3 * i + 2];

  for (int s = 0; s < PER_B; s += 256) {
    int j = base + s + t;
    float dx = pos[3 * j]     - px;
    float dy = pos[3 * j + 1] - py;
    float dz = pos[3 * j + 2] - pz;
    float d2 = dx * dx + dy * dy + dz * dz;
    if (d2 <= R2_VAL) {
      int slot = atomicAdd(&scount, 1);
      if (slot < CAP) {
        keys[slot] = ((unsigned long long)__float_as_uint(d2) << 32) |
                     (unsigned long long)(unsigned)j;
      }
    }
  }
  __syncthreads();

  int n = scount; if (n > CAP) n = CAP;

  if (n > KNBR) {  // block-uniform branch
    for (int e = n + t; e < CAP; e += 256) keys[e] = ~0ULL;
    __syncthreads();
    for (int k = 2; k <= CAP; k <<= 1) {
      for (int jj = k >> 1; jj > 0; jj >>= 1) {
        for (int e = t; e < CAP; e += 256) {
          int p = e ^ jj;
          if (p > e) {
            bool up = ((e & k) == 0);
            unsigned long long a = keys[e], b = keys[p];
            if (up ? (a > b) : (a < b)) { keys[e] = b; keys[p] = a; }
          }
        }
        __syncthreads();
      }
    }
    n = KNBR;
  }

  if (t < KNBR) {
    int v = (t < n) ? (int)(unsigned)(keys[t] & 0xFFFFFFFFULL) : 0;
    nbr[i * KNBR + t] = v;
  }
  if (t == 0) cnt[i] = n;
}

// ---------------------------------------------------------------------------
// Phase 2: per point, MLP over up to 64 neighbors via WMMA f16, max-aggregate.
// Block = 128 threads = 4 waves; wave w owns H1 cols [16w,16w+16) and
// H2 cols [32w,32w+32). B fragments come pre-swizzled from global (L2 hits);
// A fragments are two ds_load_b128 each.
// ---------------------------------------------------------------------------
__global__ __launch_bounds__(128) void pointconv_wmma(
    const float* __restrict__ x, const float* __restrict__ pos,
    const int* __restrict__ nbr, const int* __restrict__ cnt,
    const unsigned* __restrict__ W1x, const unsigned* __restrict__ W2x,
    const float* __restrict__ b1, const float* __restrict__ b2,
    float* __restrict__ out) {
  __shared__ __attribute__((aligned(16))) _Float16 As [64 * 32];  // K padded to 32
  __shared__ __attribute__((aligned(16))) _Float16 H1s[64 * 64];

  int i = blockIdx.x;
  int t = threadIdx.x;
  int n = cnt[i];

  __builtin_prefetch(W1x, 0, 3);   // global_prefetch_b8: warm L2 for fragments
  __builtin_prefetch(W2x, 0, 3);

  // build A rows: concat(x_j, pos_j - pos_i), zero-pad invalid rows & K tail
  {
    int r = t >> 1, half = t & 1;
    int j = (r < n) ? nbr[i * KNBR + r] : -1;
    unsigned* rowd = (unsigned*)&As[r * 32];  // 16 dwords per row
    if (half == 0) {
      if (j >= 0) {
        for (int c = 0; c < 8; ++c)
          rowd[c] = pk(x[j * C_IN + 2 * c], x[j * C_IN + 2 * c + 1]);
      } else {
        for (int c = 0; c < 8; ++c) rowd[c] = 0u;
      }
    } else {
      float rx = 0.f, ry = 0.f, rz = 0.f;
      if (j >= 0) {
        rx = pos[3 * j]     - pos[3 * i];
        ry = pos[3 * j + 1] - pos[3 * i + 1];
        rz = pos[3 * j + 2] - pos[3 * i + 2];
      }
      rowd[8] = pk(rx, ry);
      rowd[9] = pk(rz, 0.f);
      for (int c = 10; c < 16; ++c) rowd[c] = 0u;
    }
  }
  __syncthreads();

  int wave = t >> 5;
  int lane = t & 31;
  int hi   = lane >> 4;
  int l16  = lane & 15;

  typedef union { v16h h; uint4 q[2]; } frag;

  // ---- GEMM1: H1 = relu(A[64x32] @ W1[32x64] + b1) ----
  {
    int nb = wave * 16;
    frag Bf;
    const uint4* wq = (const uint4*)W1x + wave * 64 + lane * 2;
    Bf.q[0] = wq[0]; Bf.q[1] = wq[1];
    float bias = b1[nb + l16];
    for (int mt = 0; mt < 4; ++mt) {
      frag Af;
      const uint4* rq = (const uint4*)&As[(mt * 16 + l16) * 32];  // 4 uint4/row
      Af.q[0] = rq[hi];       // K pairs 0..7 (hi=0) / 8..15 (hi=1)
      Af.q[1] = rq[2 + hi];   // K pairs 16..23 / 24..31
      v8f acc;
      for (int v = 0; v < 8; ++v) acc[v] = bias;
      acc = __builtin_amdgcn_wmma_f32_16x16x32_f16(false, Af.h, false, Bf.h,
                                                   (short)0, acc, false, false);
      for (int v = 0; v < 8; ++v) {   // D: lane=N col, VGPR v -> row v+8*hi
        float r = acc[v] > 0.f ? acc[v] : 0.f;
        H1s[(mt * 16 + v + 8 * hi) * 64 + nb + l16] = (_Float16)r;
      }
    }
  }
  __syncthreads();

  // ---- GEMM2: H2 = relu(H1[64x64] @ W2[64x128] + b2); masked max over rows ----
  for (int sub = 0; sub < 2; ++sub) {
    int nt = wave * 2 + sub;
    int nb = nt * 16;
    frag B0, B1;
    const uint4* w0 = (const uint4*)W2x + (nt * 2 + 0) * 64 + lane * 2;
    const uint4* w1 = (const uint4*)W2x + (nt * 2 + 1) * 64 + lane * 2;
    B0.q[0] = w0[0]; B0.q[1] = w0[1];
    B1.q[0] = w1[0]; B1.q[1] = w1[1];
    float bias = b2[nb + l16];
    float m = -3.0e38f;
    for (int mt = 0; mt < 4; ++mt) {
      v8f acc;
      for (int v = 0; v < 8; ++v) acc[v] = bias;
      frag Af;
      const uint4* rq = (const uint4*)&H1s[(mt * 16 + l16) * 64]; // 8 uint4/row
      Af.q[0] = rq[hi];       // K = 0..31
      Af.q[1] = rq[2 + hi];
      acc = __builtin_amdgcn_wmma_f32_16x16x32_f16(false, Af.h, false, B0.h,
                                                   (short)0, acc, false, false);
      Af.q[0] = rq[4 + hi];   // K = 32..63
      Af.q[1] = rq[6 + hi];
      acc = __builtin_amdgcn_wmma_f32_16x16x32_f16(false, Af.h, false, B1.h,
                                                   (short)0, acc, false, false);
      for (int v = 0; v < 8; ++v) {
        int row = mt * 16 + v + 8 * hi;
        float r = acc[v] > 0.f ? acc[v] : 0.f;
        if (row < n && r > m) m = r;   // mask padded neighbor rows
      }
    }
    float other = __shfl_xor(m, 16, 32);   // merge lane halves (same column)
    m = m > other ? m : other;
    if (hi == 0) out[i * H2_DIM + nb + l16] = m;
  }
}

// ---------------------------------------------------------------------------
// Phase 3: tuple tail — copy pos (f32) and batch (raw bits) after out.
// ---------------------------------------------------------------------------
__global__ void copy_tail(const float* __restrict__ pos,
                          const int* __restrict__ batch,
                          float* __restrict__ out) {
  int t = blockIdx.x * blockDim.x + threadIdx.x;
  if (t < N_PTS * 3) {
    out[N_PTS * H2_DIM + t] = pos[t];
  } else {
    int u = t - N_PTS * 3;
    if (u < N_PTS)
      ((int*)out)[N_PTS * H2_DIM + N_PTS * 3 + u] = batch[u];
  }
}

// ---------------------------------------------------------------------------
extern "C" void kernel_launch(void* const* d_in, const int* in_sizes, int n_in,
                              void* d_out, int out_size, void* d_ws, size_t ws_size,
                              hipStream_t stream) {
  const float* x     = (const float*)d_in[0];
  const float* pos   = (const float*)d_in[1];
  const int*   batch = (const int*)d_in[2];
  const float* W1    = (const float*)d_in[3];
  const float* b1    = (const float*)d_in[4];
  const float* W2    = (const float*)d_in[5];
  const float* b2    = (const float*)d_in[6];
  float* out = (float*)d_out;

  char* ws = (char*)d_ws;
  int*      nbr = (int*)ws;                                       // 8192*64 ints
  int*      cnt = (int*)(ws + (size_t)N_PTS * KNBR * 4);          // 8192 ints
  unsigned* W1x = (unsigned*)(ws + (size_t)N_PTS * KNBR * 4 + (size_t)N_PTS * 4);
  unsigned* W2x = W1x + 4 * 256;

  prep_weights<<<20, 256, 0, stream>>>(W1, W2, W1x, W2x);
  radius_topk<<<N_PTS, 256, 0, stream>>>(pos, nbr, cnt);
  pointconv_wmma<<<N_PTS, 128, 0, stream>>>(x, pos, nbr, cnt, W1x, W2x, b1, b2, out);
  copy_tail<<<(N_PTS * 4 + 255) / 256, 256, 0, stream>>>(pos, batch, out);
}
